// Decoder_65678639890790
// MI455X (gfx1250) — matrix operand order repaired and checked by reading, not compiled, not run on hardware
//
#include <hip/hip_runtime.h>
#include <hip/hip_bf16.h>

// CDNA5 / gfx1250 decoder LSTM, bf16-WMMA GEMV formulation.
// Weights packed once per launch into bf16 WMMA-A fragments (~103MB -> L2 resident).

typedef __attribute__((ext_vector_type(16))) __bf16 v16bf;
typedef __attribute__((ext_vector_type(8)))  float  v8f;

#define NBLK  64
#define NTHR  256          // 8 wave32 per block -> 512 waves total
#define EDIM  2048
#define GROWS 8192         // 4*E

__device__ inline float sigm(float x) { return 1.0f / (1.0f + __expf(-x)); }
__device__ inline float tanh_fast(float x) {
    float e = __expf(-2.0f * x);
    return (1.0f - e) / (1.0f + e);
}

// y_tile[16] = W_tile(16 x 32*nk) @ vec(32*nk), via chained v_wmma_f32_16x16x32_bf16.
// Wt: pre-swizzled A fragments, 512 bf16 per k-chunk (32 lanes x 16 halfs).
// vec replicated into all 16 B columns: lanes 0-15 take k[0:16], lanes 16-31 k[16:32]
// => every column of D equals y; lanes 0 and 16 hold y[0:8] / y[8:16] in acc[0..7].
__device__ inline v8f gemv_tile(const __bf16* __restrict__ Wt,
                                const __bf16* vec, int nk, int lane)
{
    v8f acc0 = {};
    v8f acc1 = {};
    const __bf16* wl = Wt + lane * 16;
    const int voff = (lane >> 4) << 4;   // 0 or 16
#pragma unroll 2
    for (int kc = 0; kc < nk; kc += 2) {
        v16bf a0 = *(const v16bf*)(wl + (size_t)kc * 512);
        v16bf b0 = *(const v16bf*)(vec + kc * 32 + voff);
        v16bf a1 = *(const v16bf*)(wl + (size_t)(kc + 1) * 512);
        v16bf b1 = *(const v16bf*)(vec + (kc + 1) * 32 + voff);
        acc0 = __builtin_amdgcn_wmma_f32_16x16x32_bf16(false, a0, false, b0,
                                                       (short)0, acc0, false, false);
        acc1 = __builtin_amdgcn_wmma_f32_16x16x32_bf16(false, a1, false, b1,
                                                       (short)0, acc1, false, false);
    }
    return acc0 + acc1;
}

__device__ inline void grid_barrier(unsigned* cnt, unsigned* gen, unsigned nb)
{
    __syncthreads();
    __threadfence();                         // release prior writes to device scope
    if (threadIdx.x == 0) {
        unsigned g = __hip_atomic_load(gen, __ATOMIC_ACQUIRE, __HIP_MEMORY_SCOPE_AGENT);
        unsigned a = __hip_atomic_fetch_add(cnt, 1u, __ATOMIC_ACQ_REL, __HIP_MEMORY_SCOPE_AGENT);
        if (a == nb - 1u) {
            __hip_atomic_store(cnt, 0u, __ATOMIC_RELAXED, __HIP_MEMORY_SCOPE_AGENT);
            __hip_atomic_fetch_add(gen, 1u, __ATOMIC_ACQ_REL, __HIP_MEMORY_SCOPE_AGENT);
        } else {
            while (__hip_atomic_load(gen, __ATOMIC_ACQUIRE, __HIP_MEMORY_SCOPE_AGENT) == g)
                __builtin_amdgcn_s_sleep(1);
        }
    }
    __syncthreads();
    __threadfence();                         // acquire: subsequent loads see fresh data
}

// Pack fp32 weights into bf16 WMMA A fragments, this kernel's contiguous k order (16-bit A 16x32):
//   lane<16 : row m=lane,    k = {0..15} within chunk
//   lane>=16: row m=lane-16, k = {16..31}
// dst element e: chunk = e>>9, i = e&15, lane = (e>>4)&31, kc = chunk%nk, tile = chunk/nk
__global__ void pack_gemv_weights(const float* __restrict__ srcA,
                                  const float* __restrict__ srcB,
                                  int strideA, int strideB, int splitK,
                                  __bf16* __restrict__ dst, int nk, long long nelem)
{
    long long e = (long long)blockIdx.x * blockDim.x + threadIdx.x;
    if (e >= nelem) return;
    int within = (int)(e & 511);
    long long chunkid = e >> 9;
    int i  = within & 15;
    int ln = within >> 4;
    int kc = (int)(chunkid % nk);
    long long tile = chunkid / nk;
    int m = (int)tile * 16 + (ln & 15);
    int k = kc * 32 + i + ((ln >> 4) << 4);
    float v = (k < splitK) ? srcA[(size_t)m * strideA + k]
                           : srcB[(size_t)m * strideB + (k - splitK)];
    dst[e] = (__bf16)v;
}

__global__ void decoder_init(const float* __restrict__ lat,
                             __bf16* vcat, float* c0, float* c1,
                             unsigned* cnt, unsigned* gen)
{
    int i = blockIdx.x * blockDim.x + threadIdx.x;
    if (i < EDIM) {
        __bf16 h = (__bf16)lat[i];     // h_init (both layers) = latent
        vcat[i] = h;
        vcat[EDIM + i] = h;
        c0[i] = 0.0f;
        c1[i] = 0.0f;
    }
    if (i == 0) { *cnt = 0u; *gen = 0u; }
}

__global__ __launch_bounds__(NTHR, 1)
void decoder_persistent(const float* __restrict__ Wih0,
                        const float* __restrict__ bih0, const float* __restrict__ bhh0,
                        const float* __restrict__ bih1, const float* __restrict__ bhh1,
                        const float* __restrict__ bout,
                        const __bf16* __restrict__ Whh0p,
                        const __bf16* __restrict__ Wcat1p,
                        const __bf16* __restrict__ Woutp,
                        __bf16* vcat,           // [h0(2048) | h1(2048)] bf16
                        float* xf,              // current x (prev output), fp32
                        float* c0, float* c1,   // cell states, fp32
                        float* g0buf, float* g1buf, // gate pre-activations, fp32
                        unsigned* bar_cnt, unsigned* bar_gen,
                        float* out, const int* __restrict__ Tp)
{
    const int lane = threadIdx.x & 31;
    const int w    = blockIdx.x * (NTHR / 32) + (threadIdx.x >> 5); // 0..511
    const int gt   = blockIdx.x * NTHR + threadIdx.x;               // 0..16383
    const int T    = *Tp;

    for (int t = 1; t < T; ++t) {
        // ---- P1: gates0 partial = W_hh0 @ h0_{t-1};  x = out_{t-1} = W_out @ h1_{t-1}
        {
            v8f acc = gemv_tile(Whh0p + (size_t)w * (64 * 512), vcat, 64, lane);
            if ((lane & 15) == 0) {
                float* p = g0buf + w * 16 + ((lane >> 4) << 3);
#pragma unroll
                for (int r = 0; r < 8; ++r) p[r] = acc[r];
            }
        }
        if ((w & 63) == 0) {                 // 8 waves, one per distinct block
            int ot = w >> 6;
            v8f acc = gemv_tile(Woutp + (size_t)ot * (64 * 512), vcat + EDIM, 64, lane);
            if ((lane & 15) == 0) {
                int m0 = ot * 16 + ((lane >> 4) << 3);
#pragma unroll
                for (int r = 0; r < 8; ++r) {
                    float v = acc[r] + bout[m0 + r];
                    out[(size_t)(t - 1) * 128 + m0 + r] = v;
                    xf[m0 + r] = v;
                }
            }
        }
        grid_barrier(bar_cnt, bar_gen, NBLK);

        // ---- P2: act0 (+ small W_ih0 @ x dot in fp32 VALU)
        if ((gt & 7) == 0) {
            int j = gt >> 3;                 // 0..2047
            int r0 = j, r1 = j + 2048, r2 = j + 4096, r3 = j + 6144;
            float s0 = g0buf[r0] + bih0[r0] + bhh0[r0];
            float s1 = g0buf[r1] + bih0[r1] + bhh0[r1];
            float s2 = g0buf[r2] + bih0[r2] + bhh0[r2];
            float s3 = g0buf[r3] + bih0[r3] + bhh0[r3];
            for (int k = 0; k < 128; ++k) {
                float xv = xf[k];
                s0 += Wih0[(size_t)r0 * 128 + k] * xv;
                s1 += Wih0[(size_t)r1 * 128 + k] * xv;
                s2 += Wih0[(size_t)r2 * 128 + k] * xv;
                s3 += Wih0[(size_t)r3 * 128 + k] * xv;
            }
            float ii = sigm(s0), ff = sigm(s1), gg = tanh_fast(s2), oo = sigm(s3);
            float c = ff * c0[j] + ii * gg;
            c0[j] = c;
            vcat[j] = (__bf16)(oo * tanh_fast(c));       // h0_t
        }
        grid_barrier(bar_cnt, bar_gen, NBLK);

        // ---- P3: gates1 = [W_ih1|W_hh1] @ [h0_t ; h1_{t-1}]  (K = 4096)
        {
            v8f acc = gemv_tile(Wcat1p + (size_t)w * (128 * 512), vcat, 128, lane);
            if ((lane & 15) == 0) {
                float* p = g1buf + w * 16 + ((lane >> 4) << 3);
#pragma unroll
                for (int r = 0; r < 8; ++r) p[r] = acc[r];
            }
        }
        grid_barrier(bar_cnt, bar_gen, NBLK);

        // ---- P4: act1
        if ((gt & 7) == 0) {
            int j = gt >> 3;
            int r0 = j, r1 = j + 2048, r2 = j + 4096, r3 = j + 6144;
            float s0 = g1buf[r0] + bih1[r0] + bhh1[r0];
            float s1 = g1buf[r1] + bih1[r1] + bhh1[r1];
            float s2 = g1buf[r2] + bih1[r2] + bhh1[r2];
            float s3 = g1buf[r3] + bih1[r3] + bhh1[r3];
            float ii = sigm(s0), ff = sigm(s1), gg = tanh_fast(s2), oo = sigm(s3);
            float c = ff * c1[j] + ii * gg;
            c1[j] = c;
            vcat[EDIM + j] = (__bf16)(oo * tanh_fast(c)); // h1_t
        }
        grid_barrier(bar_cnt, bar_gen, NBLK);
    }

    // ---- tail: out_{T-1} = W_out @ h1_{T-1}
    if ((w & 63) == 0) {
        int ot = w >> 6;
        v8f acc = gemv_tile(Woutp + (size_t)ot * (64 * 512), vcat + EDIM, 64, lane);
        if ((lane & 15) == 0) {
            int m0 = ot * 16 + ((lane >> 4) << 3);
#pragma unroll
            for (int r = 0; r < 8; ++r)
                out[(size_t)(T - 1) * 128 + m0 + r] = acc[r] + bout[m0 + r];
        }
    }
}

extern "C" void kernel_launch(void* const* d_in, const int* in_sizes, int n_in,
                              void* d_out, int out_size, void* d_ws, size_t ws_size,
                              hipStream_t stream)
{
    const float* lat  = (const float*)d_in[0];
    const float* Wih0 = (const float*)d_in[1];
    const float* Whh0 = (const float*)d_in[2];
    const float* bih0 = (const float*)d_in[3];
    const float* bhh0 = (const float*)d_in[4];
    const float* Wih1 = (const float*)d_in[5];
    const float* Whh1 = (const float*)d_in[6];
    const float* bih1 = (const float*)d_in[7];
    const float* bhh1 = (const float*)d_in[8];
    const float* Wout = (const float*)d_in[9];
    const float* bout = (const float*)d_in[10];
    const int*   Tp   = (const int*)d_in[11];
    float* outp = (float*)d_out;

    char* ws = (char*)d_ws;
    size_t off = 0;
    auto take = [&](size_t bytes) -> char* {
        char* p = ws + off;
        off += (bytes + 255) & ~(size_t)255;
        return p;
    };
    unsigned* bar   = (unsigned*)take(256);                    // [0]=cnt, [1]=gen
    __bf16*   vcat  = (__bf16*)  take((size_t)2 * EDIM * 2);   // [h0|h1] bf16
    float*    xf    = (float*)   take(128 * 4);
    float*    c0    = (float*)   take(EDIM * 4);
    float*    c1    = (float*)   take(EDIM * 4);
    float*    g0    = (float*)   take(GROWS * 4);
    float*    g1    = (float*)   take(GROWS * 4);
    __bf16*   Whh0p = (__bf16*)  take((size_t)GROWS * EDIM * 2);       // 33.5 MB
    __bf16*   Wcat1p= (__bf16*)  take((size_t)GROWS * 2 * EDIM * 2);   // 67 MB
    __bf16*   Woutp = (__bf16*)  take((size_t)128 * EDIM * 2);         // 0.5 MB
    if (off > ws_size) return;   // workspace too small; nothing safe to do

    decoder_init<<<16, 256, 0, stream>>>(lat, vcat, c0, c1, bar + 0, bar + 1);

    long long n0 = (long long)GROWS * EDIM;       // W_hh0 pack, nk=64
    pack_gemv_weights<<<(unsigned)((n0 + 255) / 256), 256, 0, stream>>>(
        Whh0, Whh0, EDIM, EDIM, EDIM, Whh0p, 64, n0);

    long long n1 = (long long)GROWS * 2 * EDIM;   // [W_ih1 | W_hh1] pack, nk=128
    pack_gemv_weights<<<(unsigned)((n1 + 255) / 256), 256, 0, stream>>>(
        Wih1, Whh1, EDIM, EDIM, EDIM, Wcat1p, 128, n1);

    long long n2 = (long long)128 * EDIM;         // W_out pack, nk=64
    pack_gemv_weights<<<(unsigned)((n2 + 255) / 256), 256, 0, stream>>>(
        Wout, Wout, EDIM, EDIM, EDIM, Woutp, 64, n2);

    decoder_persistent<<<NBLK, NTHR, 0, stream>>>(
        Wih0, bih0, bhh0, bih1, bhh1, bout,
        Whh0p, Wcat1p, Woutp,
        vcat, xf, c0, c1, g0, g1,
        bar + 0, bar + 1, outp, Tp);
}
